// PointWiseConvolution_88175678587634
// MI455X (gfx1250) — matrix operand order repaired and checked by reading, not compiled, not run on hardware
//
#include <hip/hip_runtime.h>
#include <hip/hip_bf16.h>

typedef _Float16 v16h __attribute__((ext_vector_type(16)));
typedef _Float16 v8h  __attribute__((ext_vector_type(8)));
typedef float    v8f  __attribute__((ext_vector_type(8)));

#define NPTS   4096
#define C_IN   64
#define C_OUT  128
#define KSH    2
#define QB     16          // KSH*8 buckets
#define KDIM   (QB * C_IN) // 1024 reduction dim
#define RADIUS 0.1f

// ---------------------------------------------------------------------------
// Kernel 1: per-center spherical-grid binning + bucket means  -> f16 [N][1024]
// One block per center point. LDS: sums[16][64] + counts[16].
// ---------------------------------------------------------------------------
__global__ __launch_bounds__(256) void binAndMean(
    const float* __restrict__ points,      // [N][3]
    const float* __restrict__ attributes,  // [N][C_IN]
    _Float16* __restrict__ means)          // [N][QB*C_IN]
{
    __shared__ float s_sum[QB * C_IN];
    __shared__ float s_cnt[QB];

    const int i   = blockIdx.x;
    const int tid = threadIdx.x;

    // zero LDS accumulators
    #pragma unroll
    for (int t = tid; t < QB * C_IN; t += 256) s_sum[t] = 0.0f;
    if (tid < QB) s_cnt[tid] = 0.0f;
    __syncthreads();

    const float xi = points[i * 3 + 0];
    const float yi = points[i * 3 + 1];
    const float zi = points[i * 3 + 2];
    const float r2    = RADIUS * RADIUS;
    const float scale = (float)KSH / RADIUS;

    for (int j = tid; j < NPTS; j += 256) {
        const float dx = points[j * 3 + 0] - xi;
        const float dy = points[j * 3 + 1] - yi;
        const float dz = points[j * 3 + 2] - zi;
        const float d2 = dx * dx + dy * dy + dz * dz;
        if (d2 < r2) {
            int shell = (int)(sqrtf(d2) * scale);
            shell = shell > (KSH - 1) ? (KSH - 1) : shell;
            const int octant = (dx >= 0.0f ? 1 : 0) + (dy >= 0.0f ? 2 : 0) + (dz >= 0.0f ? 4 : 0);
            const int q = shell * 8 + octant;
            atomicAdd(&s_cnt[q], 1.0f);
            const float4* aj = (const float4*)(attributes + j * C_IN);
            #pragma unroll
            for (int c4 = 0; c4 < C_IN / 4; ++c4) {
                const float4 v = aj[c4];
                atomicAdd(&s_sum[q * C_IN + c4 * 4 + 0], v.x);
                atomicAdd(&s_sum[q * C_IN + c4 * 4 + 1], v.y);
                atomicAdd(&s_sum[q * C_IN + c4 * 4 + 2], v.z);
                atomicAdd(&s_sum[q * C_IN + c4 * 4 + 3], v.w);
            }
        }
    }
    __syncthreads();

    // means -> f16 row, layout [n][q*C_IN + c]
    #pragma unroll
    for (int t = tid; t < QB * C_IN; t += 256) {
        const int   q   = t >> 6;
        const float cnt = s_cnt[q];
        const float val = (cnt > 0.0f) ? (s_sum[t] / cnt) : 0.0f;
        means[(size_t)i * KDIM + t] = (_Float16)val;
    }
}

// ---------------------------------------------------------------------------
// Kernel 2: pack W[o][c][q] (f32) into per-lane WMMA B fragments (f16).
// B tile is 32x16 (KxN): lanes 0-15 hold K = k0..k0+7, k0+16..k0+23 for column N = lane,
// lanes 16-31 hold K = k0+8..k0+15, k0+24..k0+31 for column N = lane-16.
// Wpack index: [ntile(8)][kchunk(32)][lane(32)][elem(16)]
// ---------------------------------------------------------------------------
__global__ __launch_bounds__(256) void packW(
    const float* __restrict__ W,       // [C_OUT][C_IN][QB]
    _Float16* __restrict__ Wpack)      // [8][32][32][16]
{
    const int g = blockIdx.x * 256 + threadIdx.x;   // 8*32*32 = 8192 threads
    if (g >= 8 * 32 * 32) return;
    const int ntile = g >> 10;
    const int kc    = (g >> 5) & 31;
    const int lane  = g & 31;
    const int ocol  = ntile * 16 + (lane & 15);

    _Float16* dst = Wpack + (size_t)g * 16;
    #pragma unroll
    for (int e = 0; e < 16; ++e) {
        const int kpos = e + ((lane >= 16) ? 8 : 0) + ((e >= 8) ? 8 : 0);
        const int k    = kc * 32 + kpos;          // global K in [0,1024)
        const int q    = k >> 6;                  // bucket
        const int c    = k & 63;                  // input channel
        dst[e] = (_Float16)W[((size_t)ocol * C_IN + c) * QB + q];
    }
}

// ---------------------------------------------------------------------------
// Kernel 3: out[4096][128] = means[4096][1024] @ Wt + b via v_wmma_f32_16x16x32_f16
// Grid: 256 blocks (16-row M tiles), 8 waves/block, wave w -> N tile w.
// ---------------------------------------------------------------------------
__global__ __launch_bounds__(256) void gemmWmma(
    const _Float16* __restrict__ A,      // means [4096][1024]
    const _Float16* __restrict__ Wpack,  // [8][32][32][16]
    const float*    __restrict__ bias,   // [128]
    float* __restrict__ out)             // [4096][128]
{
    const int lane  = threadIdx.x & 31;
    const int wave  = threadIdx.x >> 5;      // 0..7 -> N tile
    const int m0    = blockIdx.x * 16;       // M tile base row
    const int col   = lane & 15;
    const int hi    = lane >> 4;             // 0 for lanes 0-15, 1 for 16-31

    // A fragment source row and half-offset per ISA 16-bit A 16x32 layout:
    // lanes 0-15: elems 0..7 = K 0..7,  elems 8..15 = K 16..23
    // lanes16-31: elems 0..7 = K 8..15, elems 8..15 = K 24..31
    const size_t arow = (size_t)(m0 + col) * KDIM;
    const int    aoff = hi * 8;

    v8f acc = {};

    #pragma unroll 4
    for (int kc = 0; kc < 32; ++kc) {
        const int k0 = kc * 32;
        const v8h alo = *(const v8h*)(A + arow + k0 + aoff);
        const v8h ahi = *(const v8h*)(A + arow + k0 + aoff + 16);
        const v16h a  = __builtin_shufflevector(alo, ahi,
            0, 1, 2, 3, 4, 5, 6, 7, 8, 9, 10, 11, 12, 13, 14, 15);
        const v16h b = *(const v16h*)(Wpack + ((((size_t)wave * 32 + kc) * 32 + lane) << 4));
        acc = __builtin_amdgcn_wmma_f32_16x16x32_f16(
            false, a, false, b, (short)0, acc, false, false);
    }

    const int ocol = wave * 16 + col;
    const float bv = bias[ocol];
    #pragma unroll
    for (int r = 0; r < 8; ++r) {
        const int row = m0 + r + hi * 8;     // C/D layout: VGPR r -> M=r (+8 for hi lanes)
        out[(size_t)row * C_OUT + ocol] = acc[r] + bv;
    }
}

// ---------------------------------------------------------------------------
extern "C" void kernel_launch(void* const* d_in, const int* in_sizes, int n_in,
                              void* d_out, int out_size, void* d_ws, size_t ws_size,
                              hipStream_t stream) {
    const float* points     = (const float*)d_in[0];  // [4096][3]
    const float* attributes = (const float*)d_in[1];  // [4096][64]
    const float* W          = (const float*)d_in[2];  // [128][64][16]
    const float* b          = (const float*)d_in[3];  // [128]
    float* out              = (float*)d_out;          // [4096][128]

    char* ws = (char*)d_ws;
    _Float16* means = (_Float16*)ws;                             // 4096*1024*2 = 8 MB
    _Float16* Wpack = (_Float16*)(ws + (size_t)NPTS * KDIM * 2); // 8*32*32*16*2 = 256 KB

    binAndMean<<<NPTS, 256, 0, stream>>>(points, attributes, means);
    packW<<<(8 * 32 * 32 + 255) / 256, 256, 0, stream>>>(W, Wpack);
    gemmWmma<<<NPTS / 16, 256, 0, stream>>>(means, Wpack, b, out);
}
